// WindowAttention_627065225619
// MI455X (gfx1250) — compile-verified, matrix operand order checked by reading, not run on hardware
//
#include <hip/hip_runtime.h>

// ---------------------------------------------------------------------------
// Types for CDNA5 WMMA / TDM (gfx1250, wave32)
// ---------------------------------------------------------------------------
typedef __attribute__((ext_vector_type(16))) __bf16 v16bf;
typedef __attribute__((ext_vector_type(8)))  __bf16 v8bf;
typedef __attribute__((ext_vector_type(4)))  __bf16 v4bf;
typedef __attribute__((ext_vector_type(8)))  float  v8f;
typedef __attribute__((ext_vector_type(4)))  unsigned int v4u;
typedef __attribute__((ext_vector_type(8)))  int v8i;
typedef __attribute__((ext_vector_type(4)))  int v4i;

#define DEV __device__ __forceinline__

DEV v8f wmma_bf16(v16bf a, v16bf b, v8f c) {
  // (neg_a, A, neg_b, B, c_mod, C, reuse_a, reuse_b)
  return __builtin_amdgcn_wmma_f32_16x16x32_bf16(false, a, false, b, (short)0, c,
                                                 false, false);
}

// ---------------------------------------------------------------------------
// TDM: 2D tile load (global -> LDS), 6-arg builtin (clang-23 lane).
// D# group0: count=1 | lds_addr | global_addr(57b) | type=2
// D# group1: data_size=2B | tensor_dim0/1 | tile_dim0/1 | tensor_dim0_stride
// higher groups zero (tensor <= 2D). Issue from ONE wave only (EXEC ignored).
// ---------------------------------------------------------------------------
DEV void tdm_load_2d(unsigned lds_off, unsigned long long gaddr,
                     unsigned tensor_d0, unsigned tensor_d1,
                     unsigned tile_d0, unsigned tile_d1,
                     unsigned long long stride0) {
  v4u g0;
  g0.x = 1u;                                   // count=1, user mode
  g0.y = lds_off;                              // LDS byte address
  g0.z = (unsigned)gaddr;                      // global_addr[31:0]
  g0.w = ((unsigned)(gaddr >> 32) & 0x01FFFFFFu) | 0x80000000u; // [56:32] | type=2
  v8i g1;
  g1[0] = (int)(1u << 16);                                           // data_size=2B
  g1[1] = (int)((tensor_d0 & 0xFFFFu) << 16);                        // dim0[15:0]
  g1[2] = (int)((tensor_d0 >> 16) | ((tensor_d1 & 0xFFFFu) << 16));  // dim0[31:16],dim1[15:0]
  g1[3] = (int)((tensor_d1 >> 16) | ((tile_d0 & 0xFFFFu) << 16));    // dim1[31:16],tile0
  g1[4] = (int)(tile_d1 & 0xFFFFu);                                  // tile1, tile2=0
  g1[5] = (int)(unsigned)(stride0 & 0xFFFFFFFFu);                    // stride0[31:0]
  g1[6] = (int)((unsigned)(stride0 >> 32) & 0xFFFFu);                // stride0[47:32]
  g1[7] = 0;
  v4i z4 = {};
  v8i z8 = {};
  __builtin_amdgcn_tensor_load_to_lds(g0, g1, z4, z4, z8, 0);
}

DEV unsigned lds_offset_of(const void* p) {
  // generic flat LDS address: aperture in high 32 bits, offset in low 32 bits
  return (unsigned)(unsigned long long)(uintptr_t)p;
}

// WMMA bf16 fragment loader from a row-major matrix (A: lane=row; B: lane=col,
// a B column equals a row of the transposed source).
// lane l<16 -> row l, K chunks {0..7},{16..23}; lane l+16 -> K {8..15},{24..31}.
DEV v16bf load_frag(const __bf16* base, int stride, int lane) {
  const int l = lane & 15, hh = lane >> 4;
  const __bf16* p = base + l * stride + hh * 8;
  v8bf c0 = *(const v8bf*)(p);
  v8bf c1 = *(const v8bf*)(p + 16);
  v16bf r;
#pragma unroll
  for (int i = 0; i < 8; ++i) { r[i] = c0[i]; r[i + 8] = c1[i]; }
  return r;
}

// Same, but rows >= rows_valid read as zero (ragged 49-row tiles).
DEV v16bf load_frag_guard(const __bf16* base, int stride, int lane, int rows_valid) {
  const int l = lane & 15, hh = lane >> 4;
  if (l >= rows_valid) { v16bf z = {}; return z; }
  const __bf16* p = base + l * stride + hh * 8;
  v8bf c0 = *(const v8bf*)(p);
  v8bf c1 = *(const v8bf*)(p + 16);
  v16bf r;
#pragma unroll
  for (int i = 0; i < 8; ++i) { r[i] = c0[i]; r[i + 8] = c1[i]; }
  return r;
}

// ---------------------------------------------------------------------------
// Problem constants
// ---------------------------------------------------------------------------
static constexpr int DIMC   = 384;
static constexpr int HEADS  = 12;
static constexpr int NTOK   = 49;
static constexpr int BWIN   = 4096;
static constexpr int HD     = 32;
static constexpr int MROWS  = BWIN * NTOK;      // 200704 (divisible by 128)
static constexpr int N_QKV  = 3 * DIMC;         // 1152 = 9 * 128
static constexpr int KSTEPS = DIMC / 32;        // 12

// Workspace layout (bytes)
static constexpr size_t OFF_QKVW = 0;                               // 1152*384 bf16
static constexpr size_t OFF_PROJW = OFF_QKVW + (size_t)1152*384*2;  // 384*384 bf16
static constexpr size_t OFF_BIAS  = OFF_PROJW + (size_t)384*384*2;  // 12*49*49 f32
static constexpr size_t OFF_Q     = ((OFF_BIAS + (size_t)12*49*49*4 + 255) / 256) * 256;
static constexpr size_t SZ_QK     = (size_t)BWIN * HEADS * NTOK * HD * 2;   // 154.1 MB
static constexpr size_t OFF_K     = OFF_Q + SZ_QK;
static constexpr size_t OFF_VT    = OFF_K + SZ_QK;
static constexpr size_t SZ_VT     = (size_t)BWIN * HEADS * HD * 64 * 2;     // 201.3 MB
static constexpr size_t OFF_AO    = OFF_VT + SZ_VT;                          // 154.1 MB

// ---------------------------------------------------------------------------
// Prep kernels
// ---------------------------------------------------------------------------
__global__ __launch_bounds__(256) void k_prep_w(const float* __restrict__ qkv_w,
                                                const float* __restrict__ proj_w,
                                                __bf16* __restrict__ wq,
                                                __bf16* __restrict__ wp) {
  int idx = blockIdx.x * 256 + threadIdx.x;
  const int nq = 1152 * 384, np = 384 * 384;
  if (idx < nq) wq[idx] = (__bf16)qkv_w[idx];
  else if (idx < nq + np) wp[idx - nq] = (__bf16)proj_w[idx - nq];
}

__global__ __launch_bounds__(256) void k_prep_bias(const float* __restrict__ bias_table,
                                                   const int* __restrict__ rel_idx,
                                                   float* __restrict__ bias_g) {
  int idx = blockIdx.x * 256 + threadIdx.x;
  if (idx >= HEADS * NTOK * NTOK) return;
  int h = idx / (NTOK * NTOK);
  int nm = idx % (NTOK * NTOK);
  bias_g[idx] = bias_table[rel_idx[nm] * HEADS + h];
}

__global__ __launch_bounds__(256) void k_zero_vt(uint4* __restrict__ vt, int n16) {
  int idx = blockIdx.x * 256 + threadIdx.x;
  if (idx < n16) vt[idx] = make_uint4(0u, 0u, 0u, 0u);
}

// ---------------------------------------------------------------------------
// QKV GEMM: (200704 x 384) @ (384 x 1152) + bias, scattered into q/k/vt tiles.
// Block = 256 threads = 8 waves, tile 128x128, BK = 32.
// B tile: TDM double-buffered (DMA for step i+1 overlaps WMMAs of step i).
// A tile: manual f32->bf16, loads batched before stores (4 outstanding b128).
// ---------------------------------------------------------------------------
__global__ __launch_bounds__(256) void k_qkv_gemm(const float* __restrict__ x,
                                                  const __bf16* __restrict__ wq,
                                                  const float* __restrict__ qkv_b,
                                                  __bf16* __restrict__ q,
                                                  __bf16* __restrict__ k,
                                                  __bf16* __restrict__ vt) {
  __shared__ __bf16 As[128 * 32];
  __shared__ __bf16 Bs[2][128 * 32];

  const int tid = threadIdx.x;
  const int wid = tid >> 5, lane = tid & 31;
  const int wm = wid & 3, wn = wid >> 2;            // 4x2 wave grid
  const int rowBase = blockIdx.x * 128;
  const int colBase = blockIdx.y * 128;
  const unsigned bs_off[2] = {lds_offset_of(&Bs[0][0]), lds_offset_of(&Bs[1][0])};
  const unsigned long long wq_base =
      (unsigned long long)(uintptr_t)wq + (size_t)colBase * DIMC * 2ull;

  v8f acc[2][4] = {};

  // Prologue: DMA first B tile
  if (wid == 0) tdm_load_2d(bs_off[0], wq_base, DIMC, N_QKV, 32, 128, DIMC);

  for (int it = 0; it < KSTEPS; ++it) {
    const int k0 = it * 32;

    // Stage A: batch the 4 global b128 loads, then convert + ds_store
    float4 fr[4];
#pragma unroll
    for (int i = 0; i < 4; ++i) {
      int v = i * 256 + tid;
      int row = v >> 3, c4 = (v & 7) * 4;
      fr[i] = *(const float4*)(x + (size_t)(rowBase + row) * DIMC + k0 + c4);
    }
#pragma unroll
    for (int i = 0; i < 4; ++i) {
      int v = i * 256 + tid;
      int row = v >> 3, c4 = (v & 7) * 4;
      v4bf o;
      o[0] = (__bf16)fr[i].x; o[1] = (__bf16)fr[i].y;
      o[2] = (__bf16)fr[i].z; o[3] = (__bf16)fr[i].w;
      *(v4bf*)(&As[row * 32 + c4]) = o;
    }

    // Pipeline next B tile; ensure current one is complete
    if (wid == 0) {
      if (it + 1 < KSTEPS) {
        tdm_load_2d(bs_off[(it + 1) & 1], wq_base + (size_t)(k0 + 32) * 2ull,
                    DIMC, N_QKV, 32, 128, DIMC);
        __builtin_amdgcn_s_wait_tensorcnt(1);   // all but the just-issued done
      } else {
        __builtin_amdgcn_s_wait_tensorcnt(0);
      }
    }
    __syncthreads();

    const __bf16* bs = &Bs[it & 1][0];
    v16bf af[2], bfv[4];
#pragma unroll
    for (int fi = 0; fi < 2; ++fi)
      af[fi] = load_frag(&As[(wm * 32 + fi * 16) * 32], 32, lane);
#pragma unroll
    for (int fj = 0; fj < 4; ++fj)
      bfv[fj] = load_frag(bs + (wn * 64 + fj * 16) * 32, 32, lane);
#pragma unroll
    for (int fi = 0; fi < 2; ++fi)
#pragma unroll
      for (int fj = 0; fj < 4; ++fj)
        acc[fi][fj] = wmma_bf16(af[fi], bfv[fj], acc[fi][fj]);
    __syncthreads();
  }

  // Epilogue: scatter into q (scaled), k, v-transposed
  const float scale = 0.17677669529663687f;   // 32^-0.5
  const int l = lane & 15, hsel = (lane >> 4) * 8;
#pragma unroll
  for (int fj = 0; fj < 4; ++fj) {
    int col = colBase + wn * 64 + fj * 16 + l;
    float qbv = qkv_b[col];
    int which = col / DIMC;
    int hh = (col % DIMC) >> 5;
    int d = col & 31;
#pragma unroll
    for (int fi = 0; fi < 2; ++fi) {
#pragma unroll
      for (int r = 0; r < 8; ++r) {
        int row = rowBase + wm * 32 + fi * 16 + r + hsel;
        int b = row / NTOK, n = row % NTOK;
        float val = acc[fi][fj][r] + qbv;
        size_t bh = (size_t)(b * HEADS + hh);
        if (which == 0)      q[(bh * NTOK + n) * HD + d] = (__bf16)(val * scale);
        else if (which == 1) k[(bh * NTOK + n) * HD + d] = (__bf16)val;
        else                 vt[(bh * HD + d) * 64 + n]  = (__bf16)val;
      }
    }
  }
}

// ---------------------------------------------------------------------------
// Fused attention: one wave per (window, head). 49x49 scores padded to 64x64,
// softmax via 16-lane shuffles, P re-staged through LDS for P@V.
// ---------------------------------------------------------------------------
__global__ __launch_bounds__(256) void k_attn(const __bf16* __restrict__ q,
                                              const __bf16* __restrict__ kk,
                                              const __bf16* __restrict__ vt,
                                              const float* __restrict__ bias_g,
                                              const float* __restrict__ mask,
                                              __bf16* __restrict__ attn_out) {
  __shared__ __bf16 P[8][64 * 64];

  const int tid = threadIdx.x;
  const int wid = tid >> 5, lane = tid & 31;
  const int g = blockIdx.x * 8 + wid;       // 0 .. 49151 = (window*12 + head)
  const int w = g / HEADS, hh = g % HEADS;
  const int win = w & 63;
  const int l = lane & 15, hsel = (lane >> 4) * 8;

  const __bf16* qb = q  + (size_t)g * NTOK * HD;
  const __bf16* kb = kk + (size_t)g * NTOK * HD;
  const __bf16* vb = vt + (size_t)g * HD * 64;

  // ---- S = q @ k^T ----
  v16bf qf[4], kf[4];
#pragma unroll
  for (int mi = 0; mi < 4; ++mi) {
    qf[mi] = load_frag_guard(qb + mi * 16 * HD, HD, lane, NTOK - mi * 16);
    kf[mi] = load_frag_guard(kb + mi * 16 * HD, HD, lane, NTOK - mi * 16);
  }
  v8f s[4][4] = {};
#pragma unroll
  for (int mi = 0; mi < 4; ++mi)
#pragma unroll
    for (int ni = 0; ni < 4; ++ni)
      s[mi][ni] = wmma_bf16(qf[mi], kf[ni], s[mi][ni]);

  // ---- + bias + mask, pad columns -> -inf ----
#pragma unroll
  for (int mi = 0; mi < 4; ++mi)
#pragma unroll
    for (int ni = 0; ni < 4; ++ni)
#pragma unroll
      for (int r = 0; r < 8; ++r) {
        int n = mi * 16 + r + hsel;
        int m = ni * 16 + l;
        float v = s[mi][ni][r];
        if (m >= NTOK) v = -1e30f;
        else if (n < NTOK)
          v += bias_g[(hh * NTOK + n) * NTOK + m] +
               mask[((size_t)win * NTOK + n) * NTOK + m];
        s[mi][ni][r] = v;
      }

  // ---- softmax over m (row lives in one 16-lane half) ----
#pragma unroll
  for (int mi = 0; mi < 4; ++mi)
#pragma unroll
    for (int r = 0; r < 8; ++r) {
      float mx = -1e30f;
#pragma unroll
      for (int ni = 0; ni < 4; ++ni) mx = fmaxf(mx, s[mi][ni][r]);
#pragma unroll
      for (int sft = 1; sft < 16; sft <<= 1) mx = fmaxf(mx, __shfl_xor(mx, sft, 32));
      float sum = 0.f;
#pragma unroll
      for (int ni = 0; ni < 4; ++ni) {
        float e = __expf(s[mi][ni][r] - mx);
        s[mi][ni][r] = e;
        sum += e;
      }
#pragma unroll
      for (int sft = 1; sft < 16; sft <<= 1) sum += __shfl_xor(sum, sft, 32);
      float inv = 1.f / sum;
#pragma unroll
      for (int ni = 0; ni < 4; ++ni) s[mi][ni][r] *= inv;
    }

  // ---- stage P (bf16) in wave-private LDS tile; DS ops are in-order per wave
#pragma unroll
  for (int mi = 0; mi < 4; ++mi)
#pragma unroll
    for (int ni = 0; ni < 4; ++ni)
#pragma unroll
      for (int r = 0; r < 8; ++r) {
        int n = mi * 16 + r + hsel;
        P[wid][n * 64 + ni * 16 + l] = (__bf16)s[mi][ni][r];
      }

  // ---- O = P @ V  (K = 64 in two 32-steps; pad cols of P are exactly 0) ----
  v8f o[4][2] = {};
#pragma unroll
  for (int ks = 0; ks < 2; ++ks) {
    v16bf pf[4], vf[2];
#pragma unroll
    for (int mi = 0; mi < 4; ++mi)
      pf[mi] = load_frag(&P[wid][mi * 16 * 64 + ks * 32], 64, lane);
#pragma unroll
    for (int nj = 0; nj < 2; ++nj)
      vf[nj] = load_frag(vb + (nj * 16) * 64 + ks * 32, 64, lane);
#pragma unroll
    for (int mi = 0; mi < 4; ++mi)
#pragma unroll
      for (int nj = 0; nj < 2; ++nj)
        o[mi][nj] = wmma_bf16(pf[mi], vf[nj], o[mi][nj]);
  }

  // ---- write (B, N, C) bf16 for proj GEMM ----
#pragma unroll
  for (int mi = 0; mi < 4; ++mi)
#pragma unroll
    for (int nj = 0; nj < 2; ++nj)
#pragma unroll
      for (int r = 0; r < 8; ++r) {
        int n = mi * 16 + r + hsel;
        if (n < NTOK) {
          int d = nj * 16 + l;
          attn_out[((size_t)w * NTOK + n) * DIMC + hh * HD + d] = (__bf16)o[mi][nj][r];
        }
      }
}

// ---------------------------------------------------------------------------
// Proj GEMM: (200704 x 384) @ (384 x 384) + bias -> f32 output.
// Both operand tiles bf16 -> both TDM double-buffered.
// ---------------------------------------------------------------------------
__global__ __launch_bounds__(256) void k_proj_gemm(const __bf16* __restrict__ a,
                                                   const __bf16* __restrict__ wp,
                                                   const float* __restrict__ proj_b,
                                                   float* __restrict__ out) {
  __shared__ __bf16 As[2][128 * 32];
  __shared__ __bf16 Bs[2][128 * 32];

  const int tid = threadIdx.x;
  const int wid = tid >> 5, lane = tid & 31;
  const int wm = wid & 3, wn = wid >> 2;
  const int rowBase = blockIdx.x * 128;
  const int colBase = blockIdx.y * 128;
  const unsigned as_off[2] = {lds_offset_of(&As[0][0]), lds_offset_of(&As[1][0])};
  const unsigned bs_off[2] = {lds_offset_of(&Bs[0][0]), lds_offset_of(&Bs[1][0])};
  const unsigned long long a_base =
      (unsigned long long)(uintptr_t)a + (size_t)rowBase * DIMC * 2ull;
  const unsigned long long b_base =
      (unsigned long long)(uintptr_t)wp + (size_t)colBase * DIMC * 2ull;

  v8f acc[2][4] = {};

  if (wid == 0) {
    tdm_load_2d(as_off[0], a_base, DIMC, MROWS, 32, 128, DIMC);
    tdm_load_2d(bs_off[0], b_base, DIMC, DIMC, 32, 128, DIMC);
  }

  for (int it = 0; it < KSTEPS; ++it) {
    if (wid == 0) {
      if (it + 1 < KSTEPS) {
        const size_t koff = (size_t)(it + 1) * 32 * 2ull;
        tdm_load_2d(as_off[(it + 1) & 1], a_base + koff, DIMC, MROWS, 32, 128, DIMC);
        tdm_load_2d(bs_off[(it + 1) & 1], b_base + koff, DIMC, DIMC, 32, 128, DIMC);
        __builtin_amdgcn_s_wait_tensorcnt(2);   // all but the 2 just-issued done
      } else {
        __builtin_amdgcn_s_wait_tensorcnt(0);
      }
    }
    __syncthreads();

    const __bf16* as = &As[it & 1][0];
    const __bf16* bs = &Bs[it & 1][0];
    v16bf af[2], bfv[4];
#pragma unroll
    for (int fi = 0; fi < 2; ++fi)
      af[fi] = load_frag(as + (wm * 32 + fi * 16) * 32, 32, lane);
#pragma unroll
    for (int fj = 0; fj < 4; ++fj)
      bfv[fj] = load_frag(bs + (wn * 64 + fj * 16) * 32, 32, lane);
#pragma unroll
    for (int fi = 0; fi < 2; ++fi)
#pragma unroll
      for (int fj = 0; fj < 4; ++fj)
        acc[fi][fj] = wmma_bf16(af[fi], bfv[fj], acc[fi][fj]);
    __syncthreads();
  }

  const int l = lane & 15, hsel = (lane >> 4) * 8;
#pragma unroll
  for (int fj = 0; fj < 4; ++fj) {
    int col = colBase + wn * 64 + fj * 16 + l;
    float pb = proj_b[col];
#pragma unroll
    for (int fi = 0; fi < 2; ++fi)
#pragma unroll
      for (int r = 0; r < 8; ++r) {
        int row = rowBase + wm * 32 + fi * 16 + r + hsel;
        out[(size_t)row * DIMC + col] = acc[fi][fj][r] + pb;
      }
  }
}

// ---------------------------------------------------------------------------
// Launcher
// ---------------------------------------------------------------------------
extern "C" void kernel_launch(void* const* d_in, const int* in_sizes, int n_in,
                              void* d_out, int out_size, void* d_ws, size_t ws_size,
                              hipStream_t stream) {
  const float* x          = (const float*)d_in[0];
  const float* mask       = (const float*)d_in[1];
  const float* qkv_w      = (const float*)d_in[2];
  const float* qkv_b      = (const float*)d_in[3];
  const float* proj_w     = (const float*)d_in[4];
  const float* proj_b     = (const float*)d_in[5];
  const float* bias_table = (const float*)d_in[6];
  const int*   rel_idx    = (const int*)d_in[7];
  float* out = (float*)d_out;

  char* ws = (char*)d_ws;
  __bf16* wq_bf  = (__bf16*)(ws + OFF_QKVW);
  __bf16* wp_bf  = (__bf16*)(ws + OFF_PROJW);
  float*  bias_g = (float*)(ws + OFF_BIAS);
  __bf16* qb     = (__bf16*)(ws + OFF_Q);
  __bf16* kb     = (__bf16*)(ws + OFF_K);
  __bf16* vt     = (__bf16*)(ws + OFF_VT);
  __bf16* ao     = (__bf16*)(ws + OFF_AO);

  // prep
  k_prep_w<<<(1152 * 384 + 384 * 384) / 256, 256, 0, stream>>>(qkv_w, proj_w, wq_bf, wp_bf);
  k_prep_bias<<<(HEADS * NTOK * NTOK + 255) / 256, 256, 0, stream>>>(bias_table, rel_idx, bias_g);
  int vt16 = (int)(SZ_VT / 16);
  k_zero_vt<<<(vt16 + 255) / 256, 256, 0, stream>>>((uint4*)vt, vt16);

  // qkv projection
  dim3 gq(MROWS / 128, N_QKV / 128);
  k_qkv_gemm<<<gq, 256, 0, stream>>>(x, wq_bf, qkv_b, qb, kb, vt);

  // fused attention (one wave per window-head)
  k_attn<<<(BWIN * HEADS) / 8, 256, 0, stream>>>(qb, kb, vt, bias_g, mask, ao);

  // output projection
  dim3 gp(MROWS / 128, DIMC / 128);
  k_proj_gemm<<<gp, 256, 0, stream>>>(ao, wp_bf, proj_b, out);
}
